// TimeSeriesModel_NStep_Combined_44968307589366
// MI455X (gfx1250) — matrix operand order repaired and checked by reading, not compiled, not run on hardware
//
#include <hip/hip_runtime.h>
#include <math.h>

// ---------------------------------------------------------------------------
// TimeSeriesModel NStep: EMA x2 -> encoder GRU (T=512) -> decoder GRU (16) -> FC
// B=128, F=8, He=Hd=512, STEPS=16, NUM_CLASSES=2, ALPHA=0.3, BETA=0.2
//
// Strategy (MI455X / gfx1250, wave32):
//  * kernel 1: EMA scans (1 thread per (b,f)), zero h buffers, init out=fc_b,
//              reset grid barrier.
//  * kernel 2: persistent, 32 blocks x 256 threads. Block c owns output
//              columns [16c,16c+16) of each GRU gate; its 3x16x512 bf16 weight
//              slice lives in LDS (48KB). Each of 8 waves owns one 16-row
//              batch tile. Per step: 16 K-chunks x 3 gates of
//              v_wmma_f32_16x16x32_bf16, then VALU/TRANS gate math, write h
//              (bf16, ping-pong in d_ws), grid-sync via atomic counter.
// ---------------------------------------------------------------------------

#define NB      128   // batch
#define NT      512   // encoder time steps
#define NF      8     // input features
#define NH      512   // hidden (enc & dec)
#define NSTEPS  16
#define NBLK    32    // NH/16 column slices
#define NTHR    256   // 8 waves

typedef __attribute__((ext_vector_type(16))) __bf16 v16bf;
typedef __attribute__((ext_vector_type(8)))  float  v8f;
typedef __attribute__((ext_vector_type(4)))  unsigned int u32x4;

union FragU { u32x4 q[2]; v16bf v; };

__device__ __forceinline__ unsigned short f2bf(float f) {
  unsigned u = __float_as_uint(f);
  unsigned r = u + 0x7FFFu + ((u >> 16) & 1u);   // round-to-nearest-even
  return (unsigned short)(r >> 16);
}
__device__ __forceinline__ float bf2f(unsigned short h) {
  return __uint_as_float(((unsigned)h) << 16);
}
__device__ __forceinline__ float fsigmoid(float x) {
  return 0.5f + 0.5f * tanhf(0.5f * x);          // TRANS v_tanh path
}

// ---------------------------------------------------------------------------
// Kernel 1: EMA chains + buffer init. 4 blocks x 256 threads (1024 = B*F).
// ---------------------------------------------------------------------------
__global__ void __launch_bounds__(NTHR) ema_init_kernel(
    const float* __restrict__ x, const float* __restrict__ fc_b,
    float* __restrict__ x2, float* __restrict__ st11, float* __restrict__ st12,
    unsigned* __restrict__ bar, unsigned short* __restrict__ hA,
    unsigned short* __restrict__ hC, float* __restrict__ out)
{
  int tid = blockIdx.x * NTHR + threadIdx.x;     // 0..1023
  if (tid == 0) *bar = 0u;
  // zero encoder h0 (hA) and decoder h0 (hC): 128*512 bf16 each
  for (int i = tid; i < NB * NH; i += 1024) { hA[i] = 0; hC[i] = 0; }
  // init out accumulator with fc bias
  for (int i = tid; i < NB * 2 * NSTEPS; i += 1024) out[i] = fc_b[i & 1];

  int b = tid >> 3, f = tid & 7;
  const float* xp  = x  + b * (NT * NF) + f;
  float*       x2p = x2 + b * (NT * NF) + f;
  float e1 = xp[0];
  float e2 = e1;
  x2p[0] = e2;
  for (int t = 1; t < NT; ++t) {
    float xv = xp[t * NF];
    e1 = 0.7f * xv + 0.3f * e1;   // ALPHA = 0.3
    e2 = 0.8f * e1 + 0.2f * e2;   // BETA  = 0.2
    x2p[t * NF] = e2;
  }
  if (f == 1 || f == 2) {
    st11[b * 2 + (f - 1)] = e1;   // x1[:, -1, [1,2]]
    st12[b * 2 + (f - 1)] = e2;   // x2[:, -1, [1,2]]
  }
}

// ---------------------------------------------------------------------------
// Shared helpers for the persistent kernel
// ---------------------------------------------------------------------------
__device__ __forceinline__ void load_wlds(const float* __restrict__ W, int c0,
                                          unsigned short* wlds) {
  // W: (3*NH, NH) row-major; slice rows g*NH + c0*16 + n -> wlds[(g*16+n)*NH+k]
  for (int i = threadIdx.x; i < 3 * 16 * NH; i += NTHR) {
    int g = i >> 13;          // / (16*512)
    int rem = i & 8191;
    int n = rem >> 9;         // / 512
    int k = rem & 511;
    wlds[i] = f2bf(W[(g * NH + c0 * 16 + n) * NH + k]);
  }
}

__device__ __forceinline__ void load_bias(const float* __restrict__ bih,
                                          const float* __restrict__ bhh, int c0,
                                          float* bih_lds, float* bhh_lds) {
  for (int i = threadIdx.x; i < 48; i += NTHR) {
    int g = i >> 4, n = i & 15;
    bih_lds[i] = bih[g * NH + c0 * 16 + n];
    bhh_lds[i] = bhh[g * NH + c0 * 16 + n];
  }
}

// D = h(16x512 bf16 tile rows) x W^T -> three 16x16 f32 tiles (r,z,n gates)
__device__ __forceinline__ void gate_matmul(const unsigned short* __restrict__ hin,
                                            const unsigned short* wlds,
                                            int m0, int lr, int hi, v8f* acc) {
  // A-frag (16x32 bf16): lane holds row m0*16+lr; lane<16 -> K {0..7,16..23},
  // lane>=16 -> K {8..15,24..31}: two b128 loads at +hi*8 and +hi*8+16.
  const unsigned short* arow = hin + (m0 * 16 + lr) * NH + hi * 8;
  // B-frag (32x16 bf16): lane holds column lr; lane<16 -> K {0..7,16..23},
  // lane>=16 -> K {8..15,24..31}.
  const unsigned short* bbase = wlds + lr * NH + hi * 8;
  for (int kc = 0; kc < NH / 32; ++kc) {
    FragU a;
    a.q[0] = *(const u32x4*)(arow + kc * 32);
    a.q[1] = *(const u32x4*)(arow + kc * 32 + 16);
#pragma unroll
    for (int g = 0; g < 3; ++g) {
      FragU b;
      const unsigned short* bp = bbase + g * 16 * NH + kc * 32;
      b.q[0] = *(const u32x4*)(bp);
      b.q[1] = *(const u32x4*)(bp + 16);
      acc[g] = __builtin_amdgcn_wmma_f32_16x16x32_bf16(
          false, a.v, false, b.v, (short)0, acc[g], false, false);
    }
  }
}

__device__ __forceinline__ void grid_sync(unsigned* bar, unsigned target) {
  __syncthreads();
  if (threadIdx.x == 0) {
    __threadfence();
    atomicAdd(bar, 1u);
    while (*(volatile unsigned*)bar < target) { __builtin_amdgcn_s_sleep(1); }
    __threadfence();
  }
  __syncthreads();
}

// ---------------------------------------------------------------------------
// Kernel 2: persistent GRU encoder + decoder + fused FC + output correction.
// Grid: 32 blocks x 256 threads (must all be resident; trivially true).
// ---------------------------------------------------------------------------
__global__ void __launch_bounds__(NTHR, 1) gru_persistent_kernel(
    const float* __restrict__ x2,
    const float* __restrict__ eWih, const float* __restrict__ eWhh,
    const float* __restrict__ ebih, const float* __restrict__ ebhh,
    const float* __restrict__ dWih, const float* __restrict__ dWhh,
    const float* __restrict__ dbih, const float* __restrict__ dbhh,
    const float* __restrict__ fcW,
    const float* __restrict__ st11, const float* __restrict__ st12,
    unsigned short* __restrict__ hA, unsigned short* __restrict__ hB,
    unsigned short* __restrict__ hC,
    unsigned* __restrict__ bar, float* __restrict__ out)
{
  const int c0   = blockIdx.x;          // column slice 0..31
  const int tid  = threadIdx.x;
  const int m0   = tid >> 5;            // wave id = batch row tile 0..7
  const int lane = tid & 31;
  const int lr   = lane & 15;
  const int hi   = lane >> 4;

  __shared__ __align__(16) unsigned short wlds[3 * 16 * NH];  // 48 KB
  __shared__ float x2_lds[NB * NF];                           // 4 KB
  __shared__ float wih_lds[3 * 16 * NF];
  __shared__ float bih_lds[48], bhh_lds[48];
  __shared__ float fcw_lds[32];

  const v8f vzero = {0.f, 0.f, 0.f, 0.f, 0.f, 0.f, 0.f, 0.f};

  // ---- phase 0: load encoder weights / biases / FC slice -------------------
  load_wlds(eWhh, c0, wlds);
  load_bias(ebih, ebhh, c0, bih_lds, bhh_lds);
  for (int i = tid; i < 3 * 16 * NF; i += NTHR) {
    int g = i >> 7, rem = i & 127, n = rem >> 3, f = rem & 7;
    wih_lds[i] = eWih[(g * NH + c0 * 16 + n) * NF + f];
  }
  for (int i = tid; i < 32; i += NTHR) {
    int c = i >> 4, n = i & 15;
    fcw_lds[i] = fcW[c * NH + c0 * 16 + n];
  }
  __syncthreads();

  unsigned round = 0;

  // ---- phase 1: encoder GRU, 512 sequential steps --------------------------
  for (int t = 0; t < NT; ++t) {
    for (int i = tid; i < NB * NF; i += NTHR)
      x2_lds[i] = x2[(i >> 3) * (NT * NF) + t * NF + (i & 7)];
    __syncthreads();

    const unsigned short* hin  = (t & 1) ? hB : hA;
    unsigned short*       hout = (t & 1) ? hA : hB;

    v8f acc[3] = {vzero, vzero, vzero};
    gate_matmul(hin, wlds, m0, lr, hi, acc);

#pragma unroll
    for (int i = 0; i < 8; ++i) {
      int row = m0 * 16 + i + 8 * hi;
      const float* xr = &x2_lds[row * NF];
      float gir = bih_lds[lr], giz = bih_lds[16 + lr], gin = bih_lds[32 + lr];
#pragma unroll
      for (int f = 0; f < NF; ++f) {
        float xv = xr[f];
        gir += xv * wih_lds[lr * NF + f];
        giz += xv * wih_lds[(16 + lr) * NF + f];
        gin += xv * wih_lds[(32 + lr) * NF + f];
      }
      float r = fsigmoid(gir + acc[0][i] + bhh_lds[lr]);
      float z = fsigmoid(giz + acc[1][i] + bhh_lds[16 + lr]);
      float n = tanhf(gin + r * (acc[2][i] + bhh_lds[32 + lr]));
      float hp = bf2f(hin[row * NH + c0 * 16 + lr]);
      hout[row * NH + c0 * 16 + lr] = f2bf((1.f - z) * n + z * hp);
    }
    grid_sync(bar, (++round) * NBLK);
  }
  // final encoder h is in hA (t=511 wrote hA)

  // ---- phase 2: decoder input projection gi = h_enc @ dWih^T + dbih --------
  __syncthreads();
  load_wlds(dWih, c0, wlds);                  // dWih is (3*NH, NH): same shape
  load_bias(dbih, dbhh, c0, bih_lds, bhh_lds);
  __syncthreads();

  v8f gid[3] = {vzero, vzero, vzero};
  gate_matmul(hA, wlds, m0, lr, hi, gid);
#pragma unroll
  for (int g = 0; g < 3; ++g)
#pragma unroll
    for (int i = 0; i < 8; ++i) gid[g][i] += bih_lds[g * 16 + lr];

  __syncthreads();
  load_wlds(dWhh, c0, wlds);
  __syncthreads();

  // ---- phase 3: decoder GRU, 16 steps, FC fused into epilogue --------------
  for (int s = 0; s < NSTEPS; ++s) {
    const unsigned short* hin  = (s & 1) ? hB : hC;
    unsigned short*       hout = (s & 1) ? hC : hB;

    v8f acc[3] = {vzero, vzero, vzero};
    gate_matmul(hin, wlds, m0, lr, hi, acc);

#pragma unroll
    for (int i = 0; i < 8; ++i) {
      int row = m0 * 16 + i + 8 * hi;
      float r = fsigmoid(gid[0][i] + acc[0][i] + bhh_lds[lr]);
      float z = fsigmoid(gid[1][i] + acc[1][i] + bhh_lds[16 + lr]);
      float n = tanhf(gid[2][i] + r * (acc[2][i] + bhh_lds[32 + lr]));
      float hp = bf2f(hin[row * NH + c0 * 16 + lr]);
      float hn = (1.f - z) * n + z * hp;
      hout[row * NH + c0 * 16 + lr] = f2bf(hn);

      // partial FC: out[row][2s+c] += hn * fcW[c][col]; reduce over 16 cols
      float p0 = hn * fcw_lds[lr];
      float p1 = hn * fcw_lds[16 + lr];
#pragma unroll
      for (int m = 8; m >= 1; m >>= 1) {
        p0 += __shfl_xor(p0, m, 16);
        p1 += __shfl_xor(p1, m, 16);
      }
      if (lr == 0) {
        atomicAdd(&out[row * (2 * NSTEPS) + s * 2 + 0], p0);
        atomicAdd(&out[row * (2 * NSTEPS) + s * 2 + 1], p1);
      }
    }
    grid_sync(bar, (++round) * NBLK);
  }

  // ---- phase 4: EMA-stat output correction (block 0 only) ------------------
  if (c0 == 0) {
    for (int i = tid; i < NB * 2 * NSTEPS; i += NTHR) {
      int b = i >> 5, c = i & 1;
      float v = out[i];
      v = (v - 0.2f * st12[b * 2 + c]) * (1.f / 0.8f);
      v = (v - 0.3f * st11[b * 2 + c]) * (1.f / 0.7f);
      out[i] = v;
    }
  }
}

// ---------------------------------------------------------------------------
// Host launcher
// ---------------------------------------------------------------------------
extern "C" void kernel_launch(void* const* d_in, const int* in_sizes, int n_in,
                              void* d_out, int out_size, void* d_ws, size_t ws_size,
                              hipStream_t stream) {
  (void)in_sizes; (void)n_in; (void)out_size; (void)ws_size;
  const float* x    = (const float*)d_in[0];
  const float* eWih = (const float*)d_in[1];
  const float* eWhh = (const float*)d_in[2];
  const float* ebih = (const float*)d_in[3];
  const float* ebhh = (const float*)d_in[4];
  const float* dWih = (const float*)d_in[5];
  const float* dWhh = (const float*)d_in[6];
  const float* dbih = (const float*)d_in[7];
  const float* dbhh = (const float*)d_in[8];
  const float* fcW  = (const float*)d_in[9];
  const float* fcb  = (const float*)d_in[10];
  float* out = (float*)d_out;

  char* ws = (char*)d_ws;
  float*          x2   = (float*)(ws);                              // 2 MB
  float*          st11 = (float*)(ws + 2097152);                    // 1 KB
  float*          st12 = (float*)(ws + 2098176);                    // 1 KB
  unsigned*       bar  = (unsigned*)(ws + 2099200);                 // 256 B
  unsigned short* hA   = (unsigned short*)(ws + 2099456);           // 128 KB
  unsigned short* hB   = (unsigned short*)(ws + 2099456 + 131072);  // 128 KB
  unsigned short* hC   = (unsigned short*)(ws + 2099456 + 262144);  // 128 KB

  ema_init_kernel<<<4, NTHR, 0, stream>>>(x, fcb, x2, st11, st12, bar, hA, hC, out);
  gru_persistent_kernel<<<NBLK, NTHR, 0, stream>>>(
      x2, eWih, eWhh, ebih, ebhh, dWih, dWhh, dbih, dbhh, fcW,
      st11, st12, hA, hB, hC, bar, out);
}
